// TransLayer_80418967650844
// MI455X (gfx1250) — matrix operand order repaired and checked by reading, not compiled, not run on hardware
//
#include <hip/hip_runtime.h>

// ---------------------------------------------------------------------------
// Nystromformer layer for MI455X (gfx1250, wave32, WMMA).
// All heavy GEMMs: v_wmma_f32_16x16x32_bf16 (f32 accum).
// Pinv iterations:  v_wmma_f32_16x16x4_f32 (fp32, fallback to FMA if absent).
// Tile staging:     GLOBAL_LOAD_ASYNC_TO_LDS_B128 + s_wait_asynccnt when the
//                   toolchain exposes the gfx1250 async builtins.
// ---------------------------------------------------------------------------

typedef __bf16 bf16;
typedef __attribute__((ext_vector_type(16))) __bf16 v16bf;
typedef __attribute__((ext_vector_type(8)))  float  v8f;
typedef __attribute__((ext_vector_type(2)))  float  v2f;
typedef int v4i_t __attribute__((__vector_size__(16)));   // matches builtin param type

#define B_C     4
#define N_C     8192
#define D_C     512
#define H_C     8
#define DH_C    64
#define M_C     256        // landmarks
#define BH_C    32         // B_C*H_C
#define ROWS_C  32768      // B_C*N_C
#define SCALE_Q 0.125f     // 64^-0.5
#define KCONV   33

#if defined(__HIP_DEVICE_COMPILE__) && __has_builtin(__builtin_amdgcn_wmma_f32_16x16x4_f32)
#define HAVE_WMMA_F32 1
#else
#define HAVE_WMMA_F32 0
#endif

#if defined(__HIP_DEVICE_COMPILE__) && __has_builtin(__builtin_amdgcn_global_load_async_to_lds_b128)
#define HAVE_ASYNC_LDS 1
#else
#define HAVE_ASYNC_LDS 0
#endif

#define ASG __attribute__((address_space(1)))
#define ASL __attribute__((address_space(3)))

// 16-byte tile copy global->LDS. Async path bypasses VGPRs (ASYNCcnt-tracked);
// fallback is a plain register round-trip.
__device__ inline void copy16_g2l(const void* gsrc, void* ldst) {
#if HAVE_ASYNC_LDS
    // generic global pointer bits == AS1 bits; generic LDS addr low 32 bits ==
    // AS3 offset (ISA aperture rule), so integer casts are exact.
    __builtin_amdgcn_global_load_async_to_lds_b128(
        (ASG v4i_t*)(unsigned long long)gsrc,
        (ASL v4i_t*)(unsigned int)(unsigned long long)ldst, 0, 0);
#else
    *(uint4*)ldst = *(const uint4*)gsrc;
#endif
}

__device__ inline void copy_wait() {
#if HAVE_ASYNC_LDS
#if __has_builtin(__builtin_amdgcn_s_wait_asynccnt)
    __builtin_amdgcn_s_wait_asynccnt(0);
#else
    asm volatile("s_wait_asynccnt 0x0" ::: "memory");
#endif
#endif
}

// ---------------- fragment loaders (ISA 7.12.2 layouts, wave32) -------------

union ABits { v16bf v; unsigned int u[8]; };

// A 16x32 bf16: lane l -> row m=l&15, K halves by l>>4.
__device__ inline v16bf load_a_frag(const bf16* lds, int pitch, int mrow, int lane) {
    ABits f;
    int m  = mrow + (lane & 15);
    int kw = (lane >> 4) * 4;                 // pair-index base (K pairs)
    const unsigned int* row = (const unsigned int*)(lds + (long long)m * pitch);
#pragma unroll
    for (int j = 0; j < 4; ++j) {
        f.u[j]     = row[kw + j];             // K = kb..kb+7
        f.u[4 + j] = row[8 + kw + j];         // K = 16+kb..16+kb+7
    }
    return f.v;
}

// B 32x16 bf16: lane l -> col n=l&15, K halves by l>>4 (same K order as A).
__device__ inline v16bf load_b_frag(const bf16* lds, int pitch, int ncol, int lane) {
    ABits f;
    int n  = ncol + (lane & 15);
    int kb = (lane >> 4) * 8;
#pragma unroll
    for (int j = 0; j < 8; ++j) {
        f.v[j]     = lds[(kb + j) * pitch + n];
        f.v[8 + j] = lds[(16 + kb + j) * pitch + n];
    }
    return f.v;
}

__device__ inline void zero_acc(v8f* a, int n) {
#pragma unroll
    for (int i = 0; i < 8; ++i)
        for (int j = 0; j < n; ++j) a[j][i] = 0.0f;
}

// ------------------------------- small kernels ------------------------------

__global__ void k_f32_to_bf16(const float* __restrict__ in, bf16* __restrict__ out,
                              long long n) {
    long long i = (long long)blockIdx.x * 256 + threadIdx.x;
    if (i < n) out[i] = (bf16)in[i];
}

__global__ __launch_bounds__(256) void k_layernorm(const float* __restrict__ x,
                                                   const float* __restrict__ gamma,
                                                   const float* __restrict__ beta,
                                                   bf16* __restrict__ xn) {
    int row = blockIdx.x, tid = threadIdx.x;
    const float* xr = x + (long long)row * D_C;
    float v0 = xr[tid], v1 = xr[tid + 256];
    __shared__ float red[256];
    red[tid] = v0 + v1; __syncthreads();
    for (int off = 128; off > 0; off >>= 1) {
        if (tid < off) red[tid] += red[tid + off];
        __syncthreads();
    }
    float mu = red[0] * (1.0f / 512.0f);
    __syncthreads();
    float d0 = v0 - mu, d1 = v1 - mu;
    red[tid] = d0 * d0 + d1 * d1; __syncthreads();
    for (int off = 128; off > 0; off >>= 1) {
        if (tid < off) red[tid] += red[tid + off];
        __syncthreads();
    }
    float rstd = rsqrtf(red[0] * (1.0f / 512.0f) + 1e-5f);
    bf16* o = xn + (long long)row * D_C;
    o[tid]       = (bf16)(d0 * rstd * gamma[tid]       + beta[tid]);
    o[tid + 256] = (bf16)(d1 * rstd * gamma[tid + 256] + beta[tid + 256]);
}

__global__ __launch_bounds__(64) void k_pool(const bf16* __restrict__ q,
                                             const bf16* __restrict__ k,
                                             bf16* __restrict__ ql, bf16* __restrict__ kl) {
    int bh = blockIdx.x >> 8, m = blockIdx.x & 255, dd = threadIdx.x;
    long long base = ((long long)bh * N_C + m * 32) * DH_C + dd;
    float sq = 0.f, sk = 0.f;
#pragma unroll 4
    for (int i = 0; i < 32; ++i) {
        sq += (float)q[base + (long long)i * DH_C];
        sk += (float)k[base + (long long)i * DH_C];
    }
    long long o = ((long long)bh * M_C + m) * DH_C + dd;
    ql[o] = (bf16)(sq * (1.f / 32.f));
    kl[o] = (bf16)(sk * (1.f / 32.f));
}

__global__ __launch_bounds__(256) void k_softmax(const float* __restrict__ S,
                                                 bf16* __restrict__ Pb,
                                                 float* __restrict__ Pf, int C) {
    long long row = blockIdx.x;
    const float* s = S + row * (long long)C;
    int tid = threadIdx.x;
    __shared__ float red[256];
    float mx = -3.4e38f;
    for (int c = tid; c < C; c += 256) mx = fmaxf(mx, s[c]);
    red[tid] = mx; __syncthreads();
    for (int off = 128; off > 0; off >>= 1) {
        if (tid < off) red[tid] = fmaxf(red[tid], red[tid + off]);
        __syncthreads();
    }
    mx = red[0]; __syncthreads();
    float sum = 0.f;
    for (int c = tid; c < C; c += 256) sum += __expf(s[c] - mx);
    red[tid] = sum; __syncthreads();
    for (int off = 128; off > 0; off >>= 1) {
        if (tid < off) red[tid] += red[tid + off];
        __syncthreads();
    }
    float inv = 1.0f / red[0];
    for (int c = tid; c < C; c += 256) {
        float p = __expf(s[c] - mx) * inv;
        if (Pb) Pb[row * (long long)C + c] = (bf16)p;
        if (Pf) Pf[row * (long long)C + c] = p;
    }
}

__global__ void k_scal_init(float* scal) {
    if (threadIdx.x < 2) ((int*)scal)[threadIdx.x] = 0;
}

// which==0: max over all (bh,i) of row-sum; which==1: of column-sum.
__global__ __launch_bounds__(256) void k_sum_max(const float* __restrict__ A2,
                                                 float* __restrict__ scal, int which) {
    long long bh = blockIdx.x >> 8;
    int i = blockIdx.x & 255, tid = threadIdx.x;
    const float* base = A2 + bh * (long long)(M_C * M_C);
    float v = (which == 0) ? fabsf(base[i * M_C + tid]) : fabsf(base[tid * M_C + i]);
    __shared__ float red[256];
    red[tid] = v; __syncthreads();
    for (int off = 128; off > 0; off >>= 1) {
        if (tid < off) red[tid] += red[tid + off];
        __syncthreads();
    }
    if (tid == 0) atomicMax((int*)scal + which, __float_as_int(red[0]));
}

__global__ void k_zinit(const float* __restrict__ A2, const float* __restrict__ scal,
                        float* __restrict__ Z) {
    long long idx = (long long)blockIdx.x * 256 + threadIdx.x;  // 32*65536 total
    long long bh = idx >> 16;
    int rem = (int)(idx & 65535), i = rem >> 8, j = rem & 255;
    float inv = 1.0f / (scal[0] * scal[1]);
    Z[bh * 65536 + i * 256 + j] = A2[bh * 65536 + j * 256 + i] * inv;
}

__global__ __launch_bounds__(256) void k_conv(const bf16* __restrict__ v,
                                              const float* __restrict__ cw,
                                              float* __restrict__ res) {
    int bh = blockIdx.x >> 11, nt = blockIdx.x & 2047;
    int nn = nt * 4 + (threadIdx.x >> 6), dd = threadIdx.x & 63;
    int h = bh & 7;
    const bf16* vb = v + (long long)bh * N_C * DH_C;
    if (nn + 64 < N_C) __builtin_prefetch(vb + (long long)(nn + 64) * DH_C + dd, 0, 0);
    float s = 0.f;
#pragma unroll
    for (int t = 0; t < KCONV; ++t) {
        int pos = nn + t - (KCONV / 2);
        if (pos >= 0 && pos < N_C)
            s += cw[h * KCONV + t] * (float)vb[(long long)pos * DH_C + dd];
    }
    res[((long long)bh * N_C + nn) * DH_C + dd] = s;
}

// ------------------------- bf16 WMMA GEMM kernels ---------------------------
// block 256 threads = 8 waves; tile 128(M) x 64(N); wave w owns rows w*16..+15.

#define APITCH 40
#define BPITCH 72

// C[M,N] (batched) = A[M,K] @ B[K,N]; optional residual add; bf16 or f32 out.
__global__ __launch_bounds__(256) void k_gemm_nn_bf16(
    const bf16* __restrict__ A, int lda, long long strideA,
    const bf16* __restrict__ B, int ldb, long long strideB,
    void* __restrict__ Cout, int ldc, long long strideC,
    const float* __restrict__ Res, long long strideRes,
    int K, int outBf16) {
    __shared__ __align__(16) bf16 As[128 * APITCH];
    __shared__ __align__(16) bf16 Bs[32 * BPITCH];
    int batch = blockIdx.z, m0 = blockIdx.x * 128, n0 = blockIdx.y * 64;
    const bf16* Ab = A + (long long)batch * strideA;
    const bf16* Bb = B + (long long)batch * strideB;
    int tid = threadIdx.x, lane = tid & 31, wave = tid >> 5;
    v8f acc[4]; zero_acc(acc, 4);
    for (int k0 = 0; k0 < K; k0 += 32) {
#pragma unroll
        for (int it = 0; it < 2; ++it) {
            int idx = tid + it * 256, r = idx >> 2, c = (idx & 3) * 8;
            copy16_g2l(Ab + (long long)(m0 + r) * lda + k0 + c, As + r * APITCH + c);
        }
        {
            int rb = tid >> 3, cb = (tid & 7) * 8;
            copy16_g2l(Bb + (long long)(k0 + rb) * ldb + n0 + cb, Bs + rb * BPITCH + cb);
        }
        copy_wait();
        __syncthreads();
        v16bf a = load_a_frag(As, APITCH, wave * 16, lane);
#pragma unroll
        for (int nc = 0; nc < 4; ++nc) {
            v16bf b = load_b_frag(Bs, BPITCH, nc * 16, lane);
            acc[nc] = __builtin_amdgcn_wmma_f32_16x16x32_bf16(
                false, a, false, b, (short)0, acc[nc], false, false);
        }
        __syncthreads();
    }
    int mb = m0 + wave * 16 + (lane >> 4) * 8, nl = lane & 15;
#pragma unroll
    for (int nc = 0; nc < 4; ++nc) {
        int n = n0 + nc * 16 + nl;
#pragma unroll
        for (int r = 0; r < 8; ++r) {
            int m = mb + r;
            long long ci = (long long)batch * strideC + (long long)m * ldc + n;
            float val = acc[nc][r];
            if (Res) val += Res[(long long)batch * strideRes + (long long)m * ldc + n];
            if (outBf16) ((bf16*)Cout)[ci] = (bf16)val;
            else         ((float*)Cout)[ci] = val;
        }
    }
}

// C[M,N] (batched, f32) = A[M,64] @ B[N,64]^T   (K fixed = 64)
__global__ __launch_bounds__(256) void k_gemm_nt_bf16(
    const bf16* __restrict__ A, long long strideA,
    const bf16* __restrict__ B, long long strideB,
    float* __restrict__ C, int ldc, long long strideC) {
    __shared__ __align__(16) bf16 As[128 * APITCH];
    __shared__ __align__(16) bf16 Bs[32 * BPITCH];
    int batch = blockIdx.z, m0 = blockIdx.x * 128, n0 = blockIdx.y * 64;
    const bf16* Ab = A + (long long)batch * strideA;
    const bf16* Bb = B + (long long)batch * strideB;
    int tid = threadIdx.x, lane = tid & 31, wave = tid >> 5;
    v8f acc[4]; zero_acc(acc, 4);
#pragma unroll
    for (int k0 = 0; k0 < 64; k0 += 32) {
#pragma unroll
        for (int it = 0; it < 2; ++it) {
            int idx = tid + it * 256, r = idx >> 2, c = (idx & 3) * 8;
            copy16_g2l(Ab + (long long)(m0 + r) * 64 + k0 + c, As + r * APITCH + c);
        }
        {   // B^T staging: 16B row fragment scattered into LDS columns
            int nn = tid >> 2, kk0 = (tid & 3) * 8;
            uint4 w = *(const uint4*)(Bb + (long long)(n0 + nn) * 64 + k0 + kk0);
            const bf16* pv = (const bf16*)&w;
#pragma unroll
            for (int j = 0; j < 8; ++j) Bs[(kk0 + j) * BPITCH + nn] = pv[j];
        }
        copy_wait();
        __syncthreads();
        v16bf a = load_a_frag(As, APITCH, wave * 16, lane);
#pragma unroll
        for (int nc = 0; nc < 4; ++nc) {
            v16bf b = load_b_frag(Bs, BPITCH, nc * 16, lane);
            acc[nc] = __builtin_amdgcn_wmma_f32_16x16x32_bf16(
                false, a, false, b, (short)0, acc[nc], false, false);
        }
        __syncthreads();
    }
    int mb = m0 + wave * 16 + (lane >> 4) * 8, nl = lane & 15;
    float* Cb = C + (long long)batch * strideC;
#pragma unroll
    for (int nc = 0; nc < 4; ++nc)
#pragma unroll
        for (int r = 0; r < 8; ++r)
            Cb[(long long)(mb + r) * ldc + n0 + nc * 16 + nl] = acc[nc][r];
}

// QKV projection: xn[32768,512] @ Wqkv[512,1536]; scatter into q(*scale)/k/v
// head-major [bh][n][64] bf16 buffers.
__global__ __launch_bounds__(256) void k_gemm_qkv(
    const bf16* __restrict__ A, const bf16* __restrict__ B,
    bf16* __restrict__ Q, bf16* __restrict__ Kd, bf16* __restrict__ V) {
    __shared__ __align__(16) bf16 As[128 * APITCH];
    __shared__ __align__(16) bf16 Bs[32 * BPITCH];
    int m0 = blockIdx.x * 128, n0 = blockIdx.y * 64;
    int tid = threadIdx.x, lane = tid & 31, wave = tid >> 5;
    v8f acc[4]; zero_acc(acc, 4);
    for (int k0 = 0; k0 < 512; k0 += 32) {
#pragma unroll
        for (int it = 0; it < 2; ++it) {
            int idx = tid + it * 256, r = idx >> 2, c = (idx & 3) * 8;
            copy16_g2l(A + (long long)(m0 + r) * 512 + k0 + c, As + r * APITCH + c);
        }
        {
            int rb = tid >> 3, cb = (tid & 7) * 8;
            copy16_g2l(B + (long long)(k0 + rb) * 1536 + n0 + cb, Bs + rb * BPITCH + cb);
        }
        copy_wait();
        __syncthreads();
        v16bf a = load_a_frag(As, APITCH, wave * 16, lane);
#pragma unroll
        for (int nc = 0; nc < 4; ++nc) {
            v16bf b = load_b_frag(Bs, BPITCH, nc * 16, lane);
            acc[nc] = __builtin_amdgcn_wmma_f32_16x16x32_bf16(
                false, a, false, b, (short)0, acc[nc], false, false);
        }
        __syncthreads();
    }
    int mb = m0 + wave * 16 + (lane >> 4) * 8, nl = lane & 15;
#pragma unroll
    for (int nc = 0; nc < 4; ++nc) {
        int n = n0 + nc * 16 + nl;
        int sel = n >> 9, rem = n & 511, head = rem >> 6, dd = rem & 63;
        bf16* dst = (sel == 0) ? Q : (sel == 1 ? Kd : V);
        float sc = (sel == 0) ? SCALE_Q : 1.0f;
#pragma unroll
        for (int r = 0; r < 8; ++r) {
            int m = mb + r, bb = m >> 13, nn = m & 8191;
            dst[(((long long)(bb * 8 + head)) * N_C + nn) * DH_C + dd] =
                (bf16)(acc[nc][r] * sc);
        }
    }
}

// Final projection: Y = heads(outh f32 -> bf16 on the fly) @ Wout + b_out + x.
__global__ __launch_bounds__(256) void k_gemm_final(
    const float* __restrict__ outh, const bf16* __restrict__ W,
    const float* __restrict__ bout, const float* __restrict__ x,
    float* __restrict__ Y) {
    __shared__ __align__(16) bf16 As[128 * APITCH];
    __shared__ __align__(16) bf16 Bs[32 * BPITCH];
    int m0 = blockIdx.x * 128, n0 = blockIdx.y * 64;
    int tid = threadIdx.x, lane = tid & 31, wave = tid >> 5;
    int bb = m0 >> 13, nseq0 = m0 & 8191;
    v8f acc[4]; zero_acc(acc, 4);
    for (int k0 = 0; k0 < 512; k0 += 32) {
        int head = k0 >> 6, dd0 = k0 & 63;
        const float* gA = outh + (((long long)(bb * 8 + head)) * N_C + nseq0) * DH_C + dd0;
#pragma unroll
        for (int it = 0; it < 2; ++it) {
            int idx = tid + it * 256, r = idx >> 2, c = (idx & 3) * 8;
            float4 f0 = *(const float4*)(gA + (long long)r * DH_C + c);
            float4 f1 = *(const float4*)(gA + (long long)r * DH_C + c + 4);
            bf16* d = As + r * APITCH + c;
            d[0] = (bf16)f0.x; d[1] = (bf16)f0.y; d[2] = (bf16)f0.z; d[3] = (bf16)f0.w;
            d[4] = (bf16)f1.x; d[5] = (bf16)f1.y; d[6] = (bf16)f1.z; d[7] = (bf16)f1.w;
        }
        {
            int rb = tid >> 3, cb = (tid & 7) * 8;
            copy16_g2l(W + (long long)(k0 + rb) * 512 + n0 + cb, Bs + rb * BPITCH + cb);
        }
        copy_wait();
        __syncthreads();
        v16bf a = load_a_frag(As, APITCH, wave * 16, lane);
#pragma unroll
        for (int nc = 0; nc < 4; ++nc) {
            v16bf b = load_b_frag(Bs, BPITCH, nc * 16, lane);
            acc[nc] = __builtin_amdgcn_wmma_f32_16x16x32_bf16(
                false, a, false, b, (short)0, acc[nc], false, false);
        }
        __syncthreads();
    }
    int mb = m0 + wave * 16 + (lane >> 4) * 8, nl = lane & 15;
#pragma unroll
    for (int nc = 0; nc < 4; ++nc) {
        int n = n0 + nc * 16 + nl;
        float bo = bout[n];
#pragma unroll
        for (int r = 0; r < 8; ++r) {
            long long idx = (long long)(mb + r) * 512 + n;
            Y[idx] = acc[nc][r] + bo + x[idx];
        }
    }
}

// ----------------- fp32 WMMA batched matmul (pinv iterations) ---------------
// C = alpha * A @ B'   with  B' = bflag ? (bcoef*I - B) : B ; all [32][256,256].
// tile 64x64, 8 waves as 4(M) x 2(N), each wave 16x32 (2 accumulators).

#define FAP 36
#define FBP 66

__global__ __launch_bounds__(256) void k_bmm_f32(
    const float* __restrict__ A, const float* __restrict__ B, float* __restrict__ C,
    float alpha, float bcoef, int bflag) {
    __shared__ __align__(16) float As[64 * FAP];
    __shared__ float Bs[32 * FBP];
    int batch = blockIdx.z, m0 = blockIdx.x * 64, n0 = blockIdx.y * 64;
    const float* Ab = A + (long long)batch * 65536;
    const float* Bb = B + (long long)batch * 65536;
    int tid = threadIdx.x, lane = tid & 31, wave = tid >> 5;
    int wm = wave & 3, wn = wave >> 2;
    v8f acc[2]; zero_acc(acc, 2);
    for (int k0 = 0; k0 < 256; k0 += 32) {
        {
            int r = tid >> 2, c = (tid & 3) * 8;
            *(float4*)(As + r * FAP + c) =
                *(const float4*)(Ab + (long long)(m0 + r) * 256 + k0 + c);
            *(float4*)(As + r * FAP + c + 4) =
                *(const float4*)(Ab + (long long)(m0 + r) * 256 + k0 + c + 4);
        }
        {
            int rb = tid >> 3, cb = (tid & 7) * 8;
#pragma unroll
            for (int j = 0; j < 8; ++j) {
                int gk = k0 + rb, gn = n0 + cb + j;
                float bv = Bb[(long long)gk * 256 + gn];
                if (bflag) bv = ((gk == gn) ? bcoef : 0.0f) - bv;
                Bs[rb * FBP + cb + j] = bv;
            }
        }
        __syncthreads();
#pragma unroll
        for (int ks = 0; ks < 32; ks += 4) {
#if HAVE_WMMA_F32
            v2f a;
            int am = wm * 16 + (lane & 15), ak = ks + (lane >> 4) * 2;
            a[0] = As[am * FAP + ak];
            a[1] = As[am * FAP + ak + 1];
#pragma unroll
            for (int nc = 0; nc < 2; ++nc) {
                v2f b;
                int bn = wn * 32 + nc * 16 + (lane & 15), bk = ks + (lane >> 4) * 2;
                b[0] = Bs[bk * FBP + bn];
                b[1] = Bs[(bk + 1) * FBP + bn];
                acc[nc] = __builtin_amdgcn_wmma_f32_16x16x4_f32(
                    false, a, false, b, (short)0, acc[nc], false, false);
            }
#else
            int ml8 = (lane >> 4) * 8, nl = lane & 15;
#pragma unroll
            for (int nc = 0; nc < 2; ++nc)
#pragma unroll
                for (int r = 0; r < 8; ++r) {
                    float s = acc[nc][r];
#pragma unroll
                    for (int kk = 0; kk < 4; ++kk)
                        s += As[(wm * 16 + r + ml8) * FAP + ks + kk] *
                             Bs[(ks + kk) * FBP + wn * 32 + nc * 16 + nl];
                    acc[nc][r] = s;
                }
#endif
        }
        __syncthreads();
    }
    int ml8 = (lane >> 4) * 8, nl = lane & 15;
    float* Cb = C + (long long)batch * 65536;
#pragma unroll
    for (int nc = 0; nc < 2; ++nc)
#pragma unroll
        for (int r = 0; r < 8; ++r) {
            int m = m0 + wm * 16 + r + ml8, n = n0 + wn * 32 + nc * 16 + nl;
            Cb[(long long)m * 256 + n] = alpha * acc[nc][r];
        }
}

// ------------------------------ orchestration -------------------------------

typedef unsigned long long u64;
static constexpr u64 OFF_WQKV  = 0;
static constexpr u64 OFF_WOUT  = OFF_WQKV  + 512ull * 1536 * 2;
static constexpr u64 OFF_XN    = OFF_WOUT  + 512ull * 512 * 2;
static constexpr u64 OFF_Q     = OFF_XN    + (u64)ROWS_C * D_C * 2;
static constexpr u64 OFF_K     = OFF_Q     + (u64)BH_C * N_C * DH_C * 2;
static constexpr u64 OFF_V     = OFF_K     + (u64)BH_C * N_C * DH_C * 2;
static constexpr u64 OFF_QL    = OFF_V     + (u64)BH_C * N_C * DH_C * 2;
static constexpr u64 OFF_KL    = OFF_QL    + (u64)BH_C * M_C * DH_C * 2;
static constexpr u64 OFF_SIMB  = OFF_KL    + (u64)BH_C * M_C * DH_C * 2;   // sim1/sim3/T (aliased)
static constexpr u64 OFF_SIM2  = OFF_SIMB  + (u64)BH_C * N_C * M_C * 4;
static constexpr u64 OFF_ATT1  = OFF_SIM2  + (u64)BH_C * M_C * M_C * 4;
static constexpr u64 OFF_ATT2F = OFF_ATT1  + (u64)BH_C * N_C * M_C * 2;
static constexpr u64 OFF_ATT3  = OFF_ATT2F + (u64)BH_C * M_C * M_C * 4;
static constexpr u64 OFF_Z0    = OFF_ATT3  + (u64)BH_C * M_C * N_C * 2;
static constexpr u64 OFF_Z1    = OFF_Z0    + (u64)BH_C * M_C * M_C * 4;
static constexpr u64 OFF_XZ    = OFF_Z1    + (u64)BH_C * M_C * M_C * 4;
static constexpr u64 OFF_P1    = OFF_XZ    + (u64)BH_C * M_C * M_C * 4;
static constexpr u64 OFF_P2    = OFF_P1    + (u64)BH_C * M_C * M_C * 4;
static constexpr u64 OFF_ZINVB = OFF_P2    + (u64)BH_C * M_C * M_C * 4;
static constexpr u64 OFF_AVB   = OFF_ZINVB + (u64)BH_C * M_C * M_C * 2;
static constexpr u64 OFF_RES   = OFF_AVB   + (u64)BH_C * M_C * DH_C * 2;
static constexpr u64 OFF_OUTH  = OFF_RES   + (u64)BH_C * N_C * DH_C * 4;
static constexpr u64 OFF_SCAL  = OFF_OUTH  + (u64)BH_C * N_C * DH_C * 4;

extern "C" void kernel_launch(void* const* d_in, const int* in_sizes, int n_in,
                              void* d_out, int out_size, void* d_ws, size_t ws_size,
                              hipStream_t stream) {
    const float* x      = (const float*)d_in[0];
    const float* gamma  = (const float*)d_in[1];
    const float* beta   = (const float*)d_in[2];
    const float* w_qkv  = (const float*)d_in[3];
    const float* w_out  = (const float*)d_in[4];
    const float* b_out  = (const float*)d_in[5];
    const float* conv_w = (const float*)d_in[6];
    float* Y = (float*)d_out;

    char* ws = (char*)d_ws;
    bf16*  wqkvb = (bf16*)(ws + OFF_WQKV);
    bf16*  woutb = (bf16*)(ws + OFF_WOUT);
    bf16*  xnb   = (bf16*)(ws + OFF_XN);
    bf16*  qb    = (bf16*)(ws + OFF_Q);
    bf16*  kb    = (bf16*)(ws + OFF_K);
    bf16*  vb    = (bf16*)(ws + OFF_V);
    bf16*  qlb   = (bf16*)(ws + OFF_QL);
    bf16*  klb   = (bf16*)(ws + OFF_KL);
    float* simb  = (float*)(ws + OFF_SIMB);
    float* sim2  = (float*)(ws + OFF_SIM2);
    bf16*  att1b = (bf16*)(ws + OFF_ATT1);
    float* att2f = (float*)(ws + OFF_ATT2F);
    bf16*  att3b = (bf16*)(ws + OFF_ATT3);
    float* Z0    = (float*)(ws + OFF_Z0);
    float* Z1    = (float*)(ws + OFF_Z1);
    float* XZ    = (float*)(ws + OFF_XZ);
    float* P1    = (float*)(ws + OFF_P1);
    float* P2    = (float*)(ws + OFF_P2);
    bf16*  zinvb = (bf16*)(ws + OFF_ZINVB);
    bf16*  avb   = (bf16*)(ws + OFF_AVB);
    float* resf  = (float*)(ws + OFF_RES);
    float* outh  = (float*)(ws + OFF_OUTH);
    float* scal  = (float*)(ws + OFF_SCAL);
    bf16*  Tb    = (bf16*)(ws + OFF_SIMB);   // alias: sim buffer free by then

    // 0) weight conversion to bf16
    k_f32_to_bf16<<<(512 * 1536 + 255) / 256, 256, 0, stream>>>(w_qkv, wqkvb, 512LL * 1536);
    k_f32_to_bf16<<<(512 * 512 + 255) / 256, 256, 0, stream>>>(w_out, woutb, 512LL * 512);

    // 1) layernorm
    k_layernorm<<<ROWS_C, 256, 0, stream>>>(x, gamma, beta, xnb);

    // 2) qkv projection (WMMA), head-major scatter with q-scale
    k_gemm_qkv<<<dim3(ROWS_C / 128, 1536 / 64, 1), 256, 0, stream>>>(xnb, wqkvb, qb, kb, vb);

    // 3) landmark pooling
    k_pool<<<BH_C * M_C, 64, 0, stream>>>(qb, kb, qlb, klb);

    // 4) sims + softmaxes
    k_gemm_nt_bf16<<<dim3(N_C / 128, M_C / 64, BH_C), 256, 0, stream>>>(
        qb, (long long)N_C * DH_C, klb, (long long)M_C * DH_C,
        simb, M_C, (long long)N_C * M_C);
    k_softmax<<<BH_C * N_C, 256, 0, stream>>>(simb, att1b, nullptr, M_C);

    k_gemm_nt_bf16<<<dim3(M_C / 128, M_C / 64, BH_C), 256, 0, stream>>>(
        qlb, (long long)M_C * DH_C, klb, (long long)M_C * DH_C,
        sim2, M_C, (long long)M_C * M_C);
    k_softmax<<<BH_C * M_C, 256, 0, stream>>>(sim2, nullptr, att2f, M_C);

    k_gemm_nt_bf16<<<dim3(M_C / 128, N_C / 64, BH_C), 256, 0, stream>>>(
        qlb, (long long)M_C * DH_C, kb, (long long)N_C * DH_C,
        simb, N_C, (long long)M_C * N_C);
    k_softmax<<<BH_C * M_C, 256, 0, stream>>>(simb, att3b, nullptr, N_C);

    // 5) Moore-Penrose pinv of att2f (fp32 WMMA)
    k_scal_init<<<1, 32, 0, stream>>>(scal);
    k_sum_max<<<BH_C * M_C, 256, 0, stream>>>(att2f, scal, 0);
    k_sum_max<<<BH_C * M_C, 256, 0, stream>>>(att2f, scal, 1);
    k_zinit<<<(BH_C * M_C * M_C) / 256, 256, 0, stream>>>(att2f, scal, Z0);
    float* Zc = Z0;
    float* Zn = Z1;
    dim3 gb(4, 4, BH_C);
    for (int it = 0; it < 6; ++it) {
        k_bmm_f32<<<gb, 256, 0, stream>>>(att2f, Zc, XZ, 1.0f, 0.0f, 0);   // XZ = X@Z
        k_bmm_f32<<<gb, 256, 0, stream>>>(XZ, XZ, P1, 1.0f, 7.0f, 1);      // XZ@(7I-XZ)
        k_bmm_f32<<<gb, 256, 0, stream>>>(XZ, P1, P2, 1.0f, 15.0f, 1);     // XZ@(15I-P1)
        k_bmm_f32<<<gb, 256, 0, stream>>>(Zc, P2, Zn, 0.25f, 13.0f, 1);    // .25*Z@(13I-P2)
        float* t = Zc; Zc = Zn; Zn = t;
    }
    k_f32_to_bf16<<<(BH_C * M_C * M_C + 255) / 256, 256, 0, stream>>>(
        Zc, zinvb, (long long)BH_C * M_C * M_C);

    // 6) av = attn3 @ v   [256,64] per bh, bf16 out
    k_gemm_nn_bf16<<<dim3(M_C / 128, 1, BH_C), 256, 0, stream>>>(
        att3b, N_C, (long long)M_C * N_C, vb, DH_C, (long long)N_C * DH_C,
        avb, DH_C, (long long)M_C * DH_C, nullptr, 0, N_C, 1);

    // 7) T = attn1 @ zinv  [8192,256] per bh, bf16 out (into aliased sim buffer)
    k_gemm_nn_bf16<<<dim3(N_C / 128, M_C / 64, BH_C), 256, 0, stream>>>(
        att1b, M_C, (long long)N_C * M_C, zinvb, M_C, (long long)M_C * M_C,
        Tb, M_C, (long long)N_C * M_C, nullptr, 0, M_C, 1);

    // 8) depthwise conv residual on v
    k_conv<<<BH_C * (N_C / 4), 256, 0, stream>>>(vb, conv_w, resf);

    // 9) outh = T @ av + res  [8192,64] per bh, f32 out
    k_gemm_nn_bf16<<<dim3(N_C / 128, 1, BH_C), 256, 0, stream>>>(
        Tb, M_C, (long long)N_C * M_C, avb, DH_C, (long long)M_C * DH_C,
        outh, DH_C, (long long)N_C * DH_C, resf, (long long)N_C * DH_C, M_C, 0);

    // 10) final projection + bias + residual
    k_gemm_final<<<dim3(ROWS_C / 128, D_C / 64, 1), 256, 0, stream>>>(
        outh, woutb, b_out, x, Y);
}
